// CACISLoss_6992206758578
// MI455X (gfx1250) — matrix operand order repaired and checked by reading, not compiled, not run on hardware
//
#include <hip/hip_runtime.h>
#include <hip/hip_bf16.h>

// ---------------------------------------------------------------------------
// CACIS loss on MI455X (gfx1250).
// scores: (128,512) f32, targets: (128,) i32, C: (512,512) f32 -> 2 scalars.
//
// Math restructuring (see analysis): M = c_b * outer(u,u) * E with
// E = exp(-C/eps) shared across batch, so Frank-Wolfe grad argmin only needs
// g = u o (E (u o alpha)); alpha tracked as v = c * vhat with O(1) updates.
// conj = -eps*(2 ln c + ln(vhat^T E vhat)) + 2*min(f). Base pass = 1 instance.
// E is carried as bf16 hi + bf16 lo; matvecs are batched 16-instance GEMMs
// via v_wmma_f32_16x16x32_bf16 (3 wmma per tile for ~fp32 accuracy).
// ---------------------------------------------------------------------------

#define KDIM   512
#define BATCH  128
#define NSLOT  16
#define FWITER 50

typedef __attribute__((ext_vector_type(16))) __bf16 v16bf;
typedef __attribute__((ext_vector_type(8)))  float  v8f;

union FragBF { v16bf v; uint4 q[2]; };

// ---------------- prep 1: column sums of C, eps --------------------------
__global__ __launch_bounds__(512) void kprep1(const float* __restrict__ C,
                                              float* __restrict__ colsum,
                                              float* __restrict__ epsp) {
  __shared__ float red[512];
  int j = threadIdx.x;
  float s = 0.0f;
  for (int i = 0; i < KDIM; ++i) s += C[i * KDIM + j];
  colsum[j] = s;
  red[j] = s;
  __syncthreads();
  for (int st = 256; st >= 1; st >>= 1) {
    if (j < st) red[j] += red[j + st];
    __syncthreads();
  }
  if (j == 0) {
    float mean = red[0] / (float)(KDIM * KDIM - KDIM);  // diag is zero
    epsp[0] = fmaxf(mean * 2.0f, 1e-8f);
  }
}

// ---------------- prep 2: E = exp(-C/eps) split into bf16 hi/lo ----------
__global__ __launch_bounds__(256) void kprep2(const float* __restrict__ C,
                                              const float* __restrict__ epsp,
                                              __bf16* __restrict__ Ehi,
                                              __bf16* __restrict__ Elo) {
  const float inv_eps = 1.0f / epsp[0];
  const int i = blockIdx.x;
  for (int j = threadIdx.x; j < KDIM; j += 256) {
    float e = expf(-C[i * KDIM + j] * inv_eps);
    __bf16 h = (__bf16)e;
    Ehi[i * KDIM + j] = h;
    Elo[i * KDIM + j] = (__bf16)(e - (float)h);
  }
}

// A-fragment (16x32 bf16): lane L: row = m + (L&15); elems[0..15]=K[ks+off..),
// contiguous, off = (L<16?0:16). Two 16B loads, row-major E.
__device__ __forceinline__ v16bf loadA(const __bf16* __restrict__ E,
                                       int m, int rA, int offA, int ks) {
  FragBF fr;
  const __bf16* base = E + (size_t)(m + rA) * KDIM;
  fr.q[0] = *reinterpret_cast<const uint4*>(base + ks + offA);
  fr.q[1] = *reinterpret_cast<const uint4*>(base + ks + 8 + offA);
  return fr.v;
}

// B-fragment (32x16 bf16): lane L: col = L&15; elems[0..15] = contiguous
// K block [ks + (L<16?0:16), +16) of that column's v-vector (stored per-slot
// contiguously in LDS).
__device__ __forceinline__ v16bf loadB(const __bf16* vrow, int offB, int ks) {
  FragBF fr;
  const uint4* p = reinterpret_cast<const uint4*>(vrow + ks + offB);
  fr.q[0] = p[0];
  fr.q[1] = p[1];
  return fr.v;
}

__device__ __forceinline__ void matvec_tiles(const __bf16* __restrict__ Ehi,
                                             const __bf16* __restrict__ Elo,
                                             const __bf16* vhiRow,
                                             const __bf16* vloRow,
                                             int m0, int rA, int offA, int offB,
                                             v8f& acc0, v8f& acc1) {
  acc0 = (v8f){0, 0, 0, 0, 0, 0, 0, 0};
  acc1 = (v8f){0, 0, 0, 0, 0, 0, 0, 0};
  for (int s = 0; s < KDIM / 32; ++s) {
    const int ks = s * 32;
    v16bf bh = loadB(vhiRow, offB, ks);
    v16bf bl = loadB(vloRow, offB, ks);
    v16bf a0h = loadA(Ehi, m0, rA, offA, ks);
    v16bf a0l = loadA(Elo, m0, rA, offA, ks);
    v16bf a1h = loadA(Ehi, m0 + 16, rA, offA, ks);
    v16bf a1l = loadA(Elo, m0 + 16, rA, offA, ks);
    // hi*hi + hi*lo + lo*hi ~ fp32-accurate product, f32 accumulate.
    acc0 = __builtin_amdgcn_wmma_f32_16x16x32_bf16(false, a0h, false, bh, (short)0, acc0, false, false);
    acc0 = __builtin_amdgcn_wmma_f32_16x16x32_bf16(false, a0h, false, bl, (short)0, acc0, false, false);
    acc0 = __builtin_amdgcn_wmma_f32_16x16x32_bf16(false, a0l, false, bh, (short)0, acc0, false, false);
    acc1 = __builtin_amdgcn_wmma_f32_16x16x32_bf16(false, a1h, false, bh, (short)0, acc1, false, false);
    acc1 = __builtin_amdgcn_wmma_f32_16x16x32_bf16(false, a1h, false, bl, (short)0, acc1, false, false);
    acc1 = __builtin_amdgcn_wmma_f32_16x16x32_bf16(false, a1l, false, bh, (short)0, acc1, false, false);
  }
}

// ---------------- main solver: 9 WGs x 16 instance-columns ----------------
// Instances 0..127 = batch rows, 128 = base pass, 129..143 = padding.
__global__ __launch_bounds__(512) void ksolve(const float* __restrict__ scores,
                                              const int* __restrict__ targets,
                                              const float* __restrict__ colsum,
                                              const float* __restrict__ epsp,
                                              const __bf16* __restrict__ Ehi,
                                              const __bf16* __restrict__ Elo,
                                              float* __restrict__ raw) {
  __shared__ float  Uf[NSLOT][KDIM];   // u~ = exp(-(f-minf)/eps), per slot
  __shared__ float  Vf[NSLOT][KDIM];   // vhat, fp32 master
  __shared__ __bf16 Vhi[NSLOT][KDIM];  // vhat bf16 hi
  __shared__ __bf16 Vlo[NSLOT][KDIM];  // vhat bf16 lo
  __shared__ float  mval[NSLOT];
  __shared__ float  fyv[NSLOT];
  __shared__ unsigned long long smin[NSLOT];
  __shared__ float  pmat[NSLOT][NSLOT];  // per-wave partial dots

  const int tid  = threadIdx.x;
  const int wave = tid >> 5;   // 0..15 (wave32)
  const int lane = tid & 31;
  const int grp  = blockIdx.x;
  const float eps = epsp[0];
  const float inv_eps = 1.0f / eps;

  const int lo16 = (lane < 16) ? 1 : 0;
  const int rA   = lane & 15;
  const int offA = lo16 ? 0 : 16;
  const int offB = lo16 ? 0 : 16;
  const int ncol = lane & 15;      // slot (column) for B/C/D fragments
  const int m0   = wave * 32;      // this wave's row stripe

  // ---- prologue: per-slot f, min, u~, v-init (alpha=1/K scale dropped) ----
  {
    const int slot = wave;
    const int inst = grp * NSLOT + slot;
    float fv[16];
    float mn = 3.4e38f;
    for (int j = 0; j < 16; ++j) {
      const int k = lane + 32 * j;
      float s = (inst < BATCH) ? scores[inst * KDIM + k]
                               : -colsum[k] * (1.0f / KDIM);  // base_scores
      float f = 0.5f * s;
      fv[j] = f;
      mn = fminf(mn, f);
    }
    for (int m = 16; m >= 1; m >>= 1) mn = fminf(mn, __shfl_xor(mn, m, 32));
    for (int j = 0; j < 16; ++j) {
      const int k = lane + 32 * j;
      float u = __expf(-(fv[j] - mn) * inv_eps);
      Uf[slot][k] = u;
      Vf[slot][k] = u;
      __bf16 h = (__bf16)u;
      Vhi[slot][k] = h;
      Vlo[slot][k] = (__bf16)(u - (float)h);
    }
    if (lane == 0) {
      mval[slot] = mn;
      smin[slot] = ~0ull;
      fyv[slot]  = (inst < BATCH) ? scores[inst * KDIM + targets[inst]] : 0.0f;
    }
  }
  __syncthreads();

  float c = 1.0f;  // running Prod(1-gamma), identical across all threads
  for (int t = 0; t < FWITER; ++t) {
    v8f acc0, acc1;
    matvec_tiles(Ehi, Elo, &Vhi[ncol][0], &Vlo[ncol][0], m0, rA, offA, offB,
                 acc0, acc1);
    // grad = u~ o (E vhat); pack (gradbits<<32)|row -> exact argmin w/ ties
    unsigned long long key = ~0ull;
#pragma unroll
    for (int r = 0; r < 8; ++r) {
      const int mr0 = m0 + r + (lo16 ? 0 : 8);
      const int mr1 = mr0 + 16;
      float g0 = acc0[r] * Uf[ncol][mr0];
      float g1 = acc1[r] * Uf[ncol][mr1];
      unsigned long long k0 =
          ((unsigned long long)__float_as_uint(g0) << 32) | (unsigned)mr0;
      unsigned long long k1 =
          ((unsigned long long)__float_as_uint(g1) << 32) | (unsigned)mr1;
      key = (k0 < key) ? k0 : key;
      key = (k1 < key) ? k1 : key;
    }
    atomicMin(&smin[ncol], key);
    __syncthreads();

    if (t == 0) {
      // gamma = 1: alpha becomes one-hot; reset vhat, c=1.
      const int slot = wave;
      const int jstar = (int)(smin[slot] & 0xffffffffu);
      for (int j = 0; j < 16; ++j) {
        const int k = lane + 32 * j;
        float v = (k == jstar) ? Uf[slot][k] : 0.0f;
        Vf[slot][k] = v;
        __bf16 h = (__bf16)v;
        Vhi[slot][k] = h;
        Vlo[slot][k] = (__bf16)(v - (float)h);
      }
      if (lane == 0) smin[slot] = ~0ull;
    } else {
      const float gamma = 2.0f / ((float)t + 2.0f);
      c *= (1.0f - gamma);
      if (tid < NSLOT) {
        const int slot = tid;
        const int jstar = (int)(smin[slot] & 0xffffffffu);
        float v = Vf[slot][jstar] + (gamma / c) * Uf[slot][jstar];
        Vf[slot][jstar] = v;
        __bf16 h = (__bf16)v;
        Vhi[slot][jstar] = h;
        Vlo[slot][jstar] = (__bf16)(v - (float)h);
        smin[slot] = ~0ull;
      }
    }
    __syncthreads();
  }

  // ---- epilogue: S = vhat^T E vhat (deterministic tree reduce) ----
  {
    v8f acc0, acc1;
    matvec_tiles(Ehi, Elo, &Vhi[ncol][0], &Vlo[ncol][0], m0, rA, offA, offB,
                 acc0, acc1);
    float part = 0.0f;
#pragma unroll
    for (int r = 0; r < 8; ++r) {
      const int mr0 = m0 + r + (lo16 ? 0 : 8);
      part += acc0[r] * Vf[ncol][mr0];
      part += acc1[r] * Vf[ncol][mr0 + 16];
    }
    part += __shfl_xor(part, 16, 32);  // combine the two lanes of each column
    if (lane < NSLOT) pmat[wave][ncol] = part;
    __syncthreads();
    if (tid < NSLOT) {
      const int slot = tid;
      const int inst = grp * NSLOT + slot;
      float S = 0.0f;
      for (int w = 0; w < NSLOT; ++w) S += pmat[w][slot];
      // conj = -eps*(2 ln c + ln S) + 2*min(f)
      float conj = -eps * (2.0f * logf(c) + logf(S)) + 2.0f * mval[slot];
      if (inst < BATCH)       raw[inst] = conj - fyv[slot];
      else if (inst == BATCH) raw[BATCH] = conj;  // base conj (batch-invariant)
    }
  }
}

// ---------------- final scalars ------------------------------------------
__global__ __launch_bounds__(128) void kfinal(const float* __restrict__ raw,
                                              const int* __restrict__ targets,
                                              const float* __restrict__ colsum,
                                              float* __restrict__ out) {
  __shared__ float sl[BATCH];
  __shared__ float sr[BATCH];
  __shared__ int   sc[BATCH];
  const int b = threadIdx.x;
  const float rawb = raw[b];
  const float conj_base = raw[BATCH];
  // base_vec = conj_base - base_scores[t] = conj_base + colmean[t]
  const float basevec = conj_base + colsum[targets[b]] * (1.0f / KDIM);
  const bool mask = basevec > 0.0f;
  sl[b] = rawb;
  sr[b] = mask ? rawb / basevec : 0.0f;
  sc[b] = mask ? 1 : 0;
  __syncthreads();
  for (int st = 64; st >= 1; st >>= 1) {
    if (b < st) { sl[b] += sl[b + st]; sr[b] += sr[b + st]; sc[b] += sc[b + st]; }
    __syncthreads();
  }
  if (b == 0) {
    out[0] = sl[0] / (float)BATCH;
    out[1] = (sc[0] > 0) ? sr[0] / (float)sc[0] : 0.0f;
  }
}

// ---------------- launcher ------------------------------------------------
extern "C" void kernel_launch(void* const* d_in, const int* in_sizes, int n_in,
                              void* d_out, int out_size, void* d_ws,
                              size_t ws_size, hipStream_t stream) {
  (void)in_sizes; (void)n_in; (void)out_size; (void)ws_size;
  const float* scores  = (const float*)d_in[0];
  const int*   targets = (const int*)d_in[1];
  const float* C       = (const float*)d_in[2];
  float* out = (float*)d_out;

  char* ws = (char*)d_ws;
  float*  epsp   = (float*)(ws + 0);
  float*  colsum = (float*)(ws + 256);            // 512 f
  float*  raw    = (float*)(ws + 4096);           // 129 f used
  __bf16* Ehi    = (__bf16*)(ws + 8192);          // 512 KB
  __bf16* Elo    = (__bf16*)(ws + 8192 + KDIM * KDIM * 2);  // 512 KB

  kprep1<<<1, 512, 0, stream>>>(C, colsum, epsp);
  kprep2<<<KDIM, 256, 0, stream>>>(C, epsp, Ehi, Elo);
  ksolve<<<9, 512, 0, stream>>>(scores, targets, colsum, epsp, Ehi, Elo, raw);
  kfinal<<<1, BATCH, 0, stream>>>(raw, targets, colsum, out);
}
